// SelfAttention_20564303413451
// MI455X (gfx1250) — matrix operand order repaired and checked by reading, not compiled, not run on hardware
//
#include <hip/hip_runtime.h>
#include <math.h>

#define BS   16
#define SEG  2048
#define DIM  512
#define TT   (SEG/16)   // 128 seq tiles

typedef __attribute__((ext_vector_type(16))) __bf16 v16bf;
typedef __attribute__((ext_vector_type(8)))  __bf16 v8bf;
typedef __attribute__((ext_vector_type(8)))  float  v8f;
typedef __attribute__((ext_vector_type(4)))  float  v4f;

// ---- WMMA fragment helpers (CDNA5 16x16x32 bf16, wave32) -------------------
// A (16x32, bf16): lane holds row = lane&15; elems 0-7 -> K = half*8..+7,
//                  elems 8-15 -> K = 16+half*8..+7  (two 16B chunks)
// B (32x16, bf16): lane holds col n = lane&15; same K order as A
//                  -> load from B^T stored row-major (two 16B chunks)
// C (16x16, f32):  lane holds col n = lane&15; VGPR v -> row m = v + 8*half

static __device__ __forceinline__ v8f wmma_bf16(v16bf a, v16bf b, v8f c) {
  return __builtin_amdgcn_wmma_f32_16x16x32_bf16(false, a, false, b,
                                                 (short)0, c, false, false);
}

static __device__ __forceinline__ v16bf load_a_bf16(const __bf16* __restrict__ base,
                                                    int ld, int lane) {
  const int row = lane & 15, half = lane >> 4;
  const __bf16* p = base + (size_t)row * ld + half * 8;
  v8bf c0 = *(const v8bf*)(p);
  v8bf c1 = *(const v8bf*)(p + 16);
  v16bf a;
#pragma unroll
  for (int i = 0; i < 8; ++i) { a[i] = c0[i]; a[8 + i] = c1[i]; }
  return a;
}

static __device__ __forceinline__ v16bf load_b_bf16(const __bf16* __restrict__ base,
                                                    int ld, int lane) {
  const int col = lane & 15, half = lane >> 4;
  const __bf16* p = base + (size_t)col * ld + half * 8;
  v8bf c0 = *(const v8bf*)(p);
  v8bf c1 = *(const v8bf*)(p + 16);
  v16bf b;
#pragma unroll
  for (int i = 0; i < 8; ++i) { b[i] = c0[i]; b[8 + i] = c1[i]; }
  return b;
}

// ---- K0a: convert audio f32 -> bf16 (one chunk of 8 per thread) ------------
__global__ void k_cvt_audio(const float* __restrict__ src, __bf16* __restrict__ dst) {
  const size_t c = (size_t)blockIdx.x * blockDim.x + threadIdx.x;  // chunk id
  const size_t N = (size_t)BS * SEG * DIM / 8;
  if (c >= N) return;
  const float* p = src + c * 8;
  v4f a0 = *(const v4f*)(p);
  v4f a1 = *(const v4f*)(p + 4);
  v8bf o;
#pragma unroll
  for (int i = 0; i < 4; ++i) { o[i] = (__bf16)a0[i]; o[4 + i] = (__bf16)a1[i]; }
  *(v8bf*)(dst + c * 8) = o;
}

// ---- K0b: transpose+convert weights: Wt[m][h][d] = bf16(W_m[d][h]) ---------
__global__ void k_prep_weights(const float* __restrict__ W0,
                               const float* __restrict__ W1,
                               const float* __restrict__ W2,
                               __bf16* __restrict__ Wt) {
  int idx = blockIdx.x * blockDim.x + threadIdx.x;
  if (idx >= 3 * DIM * DIM) return;
  int m = idx / (DIM * DIM);
  int r = idx % (DIM * DIM);
  int h = r / DIM, d = r % DIM;
  const float* W = (m == 0) ? W0 : ((m == 1) ? W1 : W2);
  Wt[idx] = (__bf16)W[d * DIM + h];
}

__global__ void k_zero(float* __restrict__ p, int n) {
  int i = blockIdx.x * blockDim.x + threadIdx.x;
  if (i < n) p[i] = 0.0f;
}

// ---- K1: projections, one wave = 16 rows x 64 cols (4 WMMA tiles / A frag) -
__global__ void __launch_bounds__(256)
k_proj(const __bf16* __restrict__ audio_bf,
       const __bf16* __restrict__ Wt,
       const float* __restrict__ b_phi,
       const float* __restrict__ b_theta,
       const float* __restrict__ b_g,
       __bf16* __restrict__ phi,
       __bf16* __restrict__ theta,
       __bf16* __restrict__ gT) {
  const int lane = threadIdx.x & 31;
  const int wid  = blockIdx.x * (blockDim.x >> 5) + (threadIdx.x >> 5);
  const int ROWT = (BS * SEG) / 16;   // 2048
  const int CG   = DIM / 64;          // 8 column groups of 64
  if (wid >= ROWT * CG * 3) return;
  const int mat = wid / (ROWT * CG);
  const int rem = wid % (ROWT * CG);
  const int rt = rem / CG, h0 = (rem % CG) * 64;
  const float* bias = (mat == 0) ? b_phi : (mat == 1 ? b_theta : b_g);
  const __bf16* wbase = Wt + (size_t)mat * DIM * DIM;
  const __bf16* A = audio_bf + (size_t)(rt * 16) * DIM;

  const int half = lane >> 4, n = lane & 15;
  v8f c[4];
#pragma unroll
  for (int j = 0; j < 4; ++j) {
    float bv = bias[h0 + j * 16 + n];
#pragma unroll
    for (int v = 0; v < 8; ++v) c[j][v] = bv;
  }

#pragma unroll 2
  for (int k = 0; k < DIM; k += 32) {
    v16bf a = load_a_bf16(A + k, DIM, lane);
#pragma unroll
    for (int j = 0; j < 4; ++j) {
      v16bf b = load_b_bf16(wbase + (size_t)(h0 + j * 16) * DIM + k, DIM, lane);
      c[j] = wmma_bf16(a, b, c[j]);
    }
  }

  const float scale = (mat == 0) ? 0.04419417382415922f : 1.0f; // 1/sqrt(512) into phi
  if (mat < 2) {
    __bf16* mp = (mat == 0 ? phi : theta) + (size_t)(rt * 16) * DIM;
#pragma unroll
    for (int j = 0; j < 4; ++j) {
      __bf16* dst = mp + h0 + j * 16 + n;
#pragma unroll
      for (int v = 0; v < 8; ++v)
        dst[(size_t)(v + 8 * half) * DIM] = (__bf16)(c[j][v] * scale);
    }
  } else {
    const int grow = rt * 16;
    const int bidx = grow / SEG, t0 = grow % SEG;
#pragma unroll
    for (int j = 0; j < 4; ++j) {
      __bf16* dst = gT + ((size_t)bidx * DIM + h0 + j * 16 + n) * SEG + t0 + 8 * half;
      v8bf o;
#pragma unroll
      for (int v = 0; v < 8; ++v) o[v] = (__bf16)c[j][v];
      *(v8bf*)dst = o;   // transposed C store: contiguous per lane
    }
  }
}

// ---- K2: sym = 0.5*(relu(phi th^T) + relu(th phi^T)); 2-wide J groups ------
__global__ void __launch_bounds__(256)
k_scores(const __bf16* __restrict__ phi,
         const __bf16* __restrict__ theta,
         float* __restrict__ scores,
         float* __restrict__ rowsum) {
  const int lane = threadIdx.x & 31;
  const int NPG = 64 * 64 + 64;   // sum_{Jg<64}(2*Jg+2) = 4160
  const int wid = blockIdx.x * (blockDim.x >> 5) + (threadIdx.x >> 5);
  const int b = blockIdx.y;
  if (wid >= NPG) return;

  // wid -> (Jg, I): start(Jg) = Jg^2 + Jg ; I in [0, 2*Jg+2)
  int Jg = (int)((-1.0f + sqrtf(1.0f + 4.0f * (float)wid)) * 0.5f);
  if (Jg < 0) Jg = 0;
  if (Jg > 63) Jg = 63;
  while (Jg > 0 && Jg * Jg + Jg > wid) --Jg;
  while ((Jg + 1) * (Jg + 1) + (Jg + 1) <= wid) ++Jg;
  const int I = wid - (Jg * Jg + Jg);
  const int J0 = 2 * Jg, J1 = J0 + 1;   // J1 >= I always; J0 >= I unless I == J1

  const __bf16* phi_b = phi   + (size_t)b * SEG * DIM;
  const __bf16* th_b  = theta + (size_t)b * SEG * DIM;
  const __bf16* pI = phi_b + (size_t)(I * 16) * DIM;
  const __bf16* tI = th_b  + (size_t)(I * 16) * DIM;

  v8f X0 = {}, X1 = {}, Y0 = {}, Y1 = {};
#pragma unroll 2
  for (int k = 0; k < DIM; k += 32) {
    v16bf aP = load_a_bf16(pI + k, DIM, lane);
    v16bf aT = load_a_bf16(tI + k, DIM, lane);
    v16bf bT0 = load_b_bf16(th_b  + (size_t)(J0 * 16) * DIM + k, DIM, lane);
    v16bf bP0 = load_b_bf16(phi_b + (size_t)(J0 * 16) * DIM + k, DIM, lane);
    v16bf bT1 = load_b_bf16(th_b  + (size_t)(J1 * 16) * DIM + k, DIM, lane);
    v16bf bP1 = load_b_bf16(phi_b + (size_t)(J1 * 16) * DIM + k, DIM, lane);
    X0 = wmma_bf16(aP, bT0, X0);   // X_J = phi_I * theta_J^T (scale baked in phi)
    Y0 = wmma_bf16(aT, bP0, Y0);   // Y_J = theta_I * phi_J^T = X_{J,I}^T
    X1 = wmma_bf16(aP, bT1, X1);
    Y1 = wmma_bf16(aT, bP1, Y1);
  }

  const int half = lane >> 4, n = lane & 15;
  float* sc_b = scores + (size_t)b * SEG * SEG;
  float* rs   = rowsum + (size_t)b * SEG;

#pragma unroll
  for (int t = 0; t < 2; ++t) {
    const int J = (t == 0) ? J0 : J1;
    if (J < I) continue;                       // wave-uniform
    v8f X = (t == 0) ? X0 : X1;
    v8f Y = (t == 0) ? Y0 : Y1;
    float s[8];
#pragma unroll
    for (int v = 0; v < 8; ++v) {
      float xv = X[v] > 0.f ? X[v] : 0.f;
      float yv = Y[v] > 0.f ? Y[v] : 0.f;
      s[v] = 0.5f * (xv + yv);
    }
    { // tile (I,J), row-major strided C store
      float* dst = sc_b + (size_t)(I * 16) * SEG + (size_t)(J * 16) + n;
#pragma unroll
      for (int v = 0; v < 8; ++v) dst[(size_t)(v + 8 * half) * SEG] = s[v];
    }
    if (I != J) { // tile (J,I) = transpose: contiguous per-lane store
      float* dst = sc_b + (size_t)(J * 16 + n) * SEG + (size_t)(I * 16) + 8 * half;
      v4f o0, o1;
#pragma unroll
      for (int v = 0; v < 4; ++v) { o0[v] = s[v]; o1[v] = s[v + 4]; }
      *(v4f*)dst = o0;
      *(v4f*)(dst + 4) = o1;
    }
    // rows of tile (I,J): butterfly across the 16 lanes of each half
#pragma unroll
    for (int v = 0; v < 8; ++v) {
      float r = s[v];
      r += __shfl_xor(r, 1);
      r += __shfl_xor(r, 2);
      r += __shfl_xor(r, 4);
      r += __shfl_xor(r, 8);
      if (n == 0) atomicAdd(&rs[I * 16 + v + 8 * half], r);
    }
    if (I != J) { // rows of tile (J,I) = column sums of (I,J)
      float csum = 0.f;
#pragma unroll
      for (int v = 0; v < 8; ++v) csum += s[v];
      csum += __shfl_xor(csum, 16);
      if (half == 0) atomicAdd(&rs[J * 16 + n], csum);
    }
  }
}

// ---- K3: stage 32 normalized bf16 score rows in LDS (128 KB, CDNA5 WGP LDS),
// then (sym/rowsum) @ g + audio.  Per k-step: 4 ds_load_b128 (A, XOR-swizzled
// conflict-free) + 4 global B frags reused across 2 row-tiles -> 8 WMMAs.
static __device__ __forceinline__ v16bf lds_load_a(const __bf16* smem,
                                                   int rt, int r, int cbase) {
  const __bf16* base = smem + ((size_t)rt << 15);       // 16*2048 elems per tile
  v8bf lo = *(const v8bf*)(base + (size_t)((r << 8) | ( cbase      ^ r)) * 8);
  v8bf hi = *(const v8bf*)(base + (size_t)((r << 8) | ((cbase + 2) ^ r)) * 8);
  v16bf a;
#pragma unroll
  for (int i = 0; i < 8; ++i) { a[i] = lo[i]; a[8 + i] = hi[i]; }
  return a;
}

__global__ void __launch_bounds__(256)
k_attn(const float* __restrict__ audio,
       const __bf16* __restrict__ gT,
       const float* __restrict__ rowsum,
       float* __restrict__ att_out,
       float* __restrict__ scores) {
  __shared__ __bf16 smem[32 * 2048];          // 128 KB (needs CDNA5 320KB WGP LDS)

  const int tid  = threadIdx.x;
  const int lane = tid & 31;
  const int wv   = tid >> 5;                  // 8 waves, each owns 64 h-columns
  const int b  = blockIdx.x / (SEG / 32);
  const int st = blockIdx.x % (SEG / 32);     // 32-row block

  float* sc_b = scores + (size_t)b * SEG * SEG + (size_t)(st * 32) * SEG; // 32x2048
  const float* rs = rowsum + (size_t)b * SEG + st * 32;
  const __bf16* g_b = gT + (size_t)b * DIM * SEG;

  // --- staging: normalize, write back to d_out, deposit bf16 in LDS --------
  for (int c = tid; c < 32 * 256; c += 256) {  // 8-elem chunks
    const int rg = c >> 8, ci = c & 255;       // rg = 0..31
    const int rt = rg >> 4, r = rg & 15;
    const float invr = 1.0f / (rs[rg] + 1e-8f);
    float* p = sc_b + (size_t)rg * SEG + ci * 8;
    v4f a0 = *(v4f*)(p);
    v4f a1 = *(v4f*)(p + 4);
#pragma unroll
    for (int e = 0; e < 4; ++e) { a0[e] *= invr; a1[e] *= invr; }
    *(v4f*)(p)     = a0;                       // normalized scores output
    *(v4f*)(p + 4) = a1;
    v8bf o;
#pragma unroll
    for (int e = 0; e < 4; ++e) { o[e] = (__bf16)a0[e]; o[4 + e] = (__bf16)a1[e]; }
    *(v8bf*)(smem + (((size_t)rt << 15) | (size_t)((r << 8) | (ci ^ r)) * 8)) = o;
  }
  __syncthreads();

  // --- GEMM: 2 row-tiles x 4 h-tiles per wave, B reused across row-tiles ---
  const int half = lane >> 4, n = lane & 15;
  const int h0 = wv * 64;
  v8f acc[2][4] = {};
  for (int k = 0; k < SEG; k += 32) {
    const int cbase = (k >> 3) + half;        // first chunk of this lane's row
    v16bf a0 = lds_load_a(smem, 0, n, cbase);
    v16bf a1 = lds_load_a(smem, 1, n, cbase);
    const __bf16* gp = g_b + (size_t)h0 * SEG + k;
    __builtin_prefetch(gp + 64, 0, 1);        // global_prefetch_b8 (next chunk)
#pragma unroll
    for (int j = 0; j < 4; ++j) {
      v16bf bj = load_b_bf16(gp + (size_t)(j * 16) * SEG, SEG, lane);
      acc[0][j] = wmma_bf16(a0, bj, acc[0][j]);
      acc[1][j] = wmma_bf16(a1, bj, acc[1][j]);
    }
  }

  const float* au = audio   + ((size_t)b * SEG + st * 32) * DIM;
  float*       ao = att_out + ((size_t)b * SEG + st * 32) * DIM;
#pragma unroll
  for (int rt = 0; rt < 2; ++rt) {
#pragma unroll
    for (int j = 0; j < 4; ++j) {
      const int h = h0 + j * 16 + n;
#pragma unroll
      for (int v = 0; v < 8; ++v) {
        const int m = rt * 16 + v + 8 * half;
        ao[(size_t)m * DIM + h] = acc[rt][j][v] + au[(size_t)m * DIM + h];
      }
    }
  }
}

// ---------------------------------------------------------------------------
extern "C" void kernel_launch(void* const* d_in, const int* in_sizes, int n_in,
                              void* d_out, int out_size, void* d_ws, size_t ws_size,
                              hipStream_t stream) {
  const float* audio   = (const float*)d_in[0];
  const float* W_phi   = (const float*)d_in[1];
  const float* b_phi   = (const float*)d_in[2];
  const float* W_theta = (const float*)d_in[3];
  const float* b_theta = (const float*)d_in[4];
  const float* W_g     = (const float*)d_in[5];
  const float* b_g     = (const float*)d_in[6];

  float* att_out    = (float*)d_out;                              // [16,2048,512]
  float* scores_out = att_out + (size_t)BS * SEG * DIM;           // [16,2048,2048]

  char* ws = (char*)d_ws;
  const size_t MAT_B = (size_t)BS * SEG * DIM * sizeof(__bf16);   // 32 MiB
  const size_t WT_B  = (size_t)3 * DIM * DIM * sizeof(__bf16);
  __bf16* phi      = (__bf16*)(ws);
  __bf16* theta    = (__bf16*)(ws + MAT_B);
  __bf16* gT       = (__bf16*)(ws + 2 * MAT_B);
  __bf16* audio_bf = (__bf16*)(ws + 3 * MAT_B);
  __bf16* Wt       = (__bf16*)(ws + 4 * MAT_B);
  float*  rowsum   = (float*) (ws + 4 * MAT_B + WT_B);

  // K0: audio bf16 convert, weight transpose/convert, rowsum clear
  {
    const size_t nchunks = (size_t)BS * SEG * DIM / 8;
    k_cvt_audio<<<(int)((nchunks + 255) / 256), 256, 0, stream>>>(audio, audio_bf);
  }
  k_prep_weights<<<(3 * DIM * DIM + 255) / 256, 256, 0, stream>>>(W_phi, W_theta, W_g, Wt);
  k_zero<<<(BS * SEG + 255) / 256, 256, 0, stream>>>(rowsum, BS * SEG);

  // K1: projections (2048 row-tiles x 8 col-groups x 3 mats, 8 waves/block)
  {
    const int waves = (BS * SEG / 16) * (DIM / 64) * 3;   // 49152
    k_proj<<<waves / 8, 256, 0, stream>>>(audio_bf, Wt, b_phi, b_theta, b_g,
                                          phi, theta, gT);
  }

  // K2: symmetric relu scores + row sums (upper-tri, 2-wide J groups)
  {
    const int NPG = 64 * 64 + 64;                         // 4160
    dim3 grid((NPG + 7) / 8, BS);
    k_scores<<<grid, 256, 0, stream>>>(phi, theta, scores_out, rowsum);
  }

  // K3: normalize (in LDS + in place) + attention GEMM + residual
  k_attn<<<BS * (SEG / 32), 256, 0, stream>>>(audio, gT, rowsum, att_out, scores_out);
}
